// CausalSelfAttention_62740882260155
// MI455X (gfx1250) — matrix operand order repaired and checked by reading, not compiled, not run on hardware
//
#include <hip/hip_runtime.h>

// ---------------------------------------------------------------------------
// CDNA5 (gfx1250) causal self-attention: bf16 WMMA, f32 accumulate,
// double-buffered async global->LDS staging (ASYNCcnt pipeline).
// B=2, T=2048, E=1024, nh=16, dh=64.
// ---------------------------------------------------------------------------

typedef __attribute__((ext_vector_type(16))) __bf16 v16bf;
typedef __attribute__((ext_vector_type(8)))  float  v8f;

#define B_DIM   2
#define T_DIM   2048
#define E_DIM   1024
#define NH      16
#define DH      64
#define M_ROWS  (B_DIM * T_DIM)       // 4096 rows
#define E3      (3 * E_DIM)           // 3072
#define MASKB   (-10000.0f)

#if defined(__AMDGCN__) && __has_builtin(__builtin_amdgcn_global_load_async_to_lds_b128)
#define HAVE_ASYNC 1
#else
#define HAVE_ASYNC 0
#endif

#if HAVE_ASYNC
typedef int v4i __attribute__((vector_size(16)));
typedef __attribute__((address_space(1))) v4i gv4i;   // global (printed __device__)
typedef __attribute__((address_space(3))) v4i sv4i;   // LDS    (printed __shared__)
__device__ __forceinline__ void async_cp16(const void* g, void* l) {
  __builtin_amdgcn_global_load_async_to_lds_b128((gv4i*)g, (sv4i*)l, 0, 0);
}
#endif

template <int N>
__device__ __forceinline__ void wait_async() {
#if HAVE_ASYNC
#if __has_builtin(__builtin_amdgcn_s_wait_asynccnt)
  __builtin_amdgcn_s_wait_asynccnt(N);
#else
  asm volatile("s_wait_asynccnt %0" ::"i"(N));
#endif
#endif
}

// copy 32 bf16 (64B) from global to LDS: async b128 x4 or sync uint4 x4
__device__ __forceinline__ void cp64B(const __bf16* __restrict__ g, __bf16* l) {
#if HAVE_ASYNC
  async_cp16(g, l);
  async_cp16(g + 8, l + 8);
  async_cp16(g + 16, l + 16);
  async_cp16(g + 24, l + 24);
#else
  reinterpret_cast<uint4*>(l)[0] = reinterpret_cast<const uint4*>(g)[0];
  reinterpret_cast<uint4*>(l)[1] = reinterpret_cast<const uint4*>(g)[1];
  reinterpret_cast<uint4*>(l)[2] = reinterpret_cast<const uint4*>(g)[2];
  reinterpret_cast<uint4*>(l)[3] = reinterpret_cast<const uint4*>(g)[3];
#endif
}
// copy 16 bf16 (32B)
__device__ __forceinline__ void cp32B(const __bf16* __restrict__ g, __bf16* l) {
#if HAVE_ASYNC
  async_cp16(g, l);
  async_cp16(g + 8, l + 8);
#else
  reinterpret_cast<uint4*>(l)[0] = reinterpret_cast<const uint4*>(g)[0];
  reinterpret_cast<uint4*>(l)[1] = reinterpret_cast<const uint4*>(g)[1];
#endif
}

__device__ __forceinline__ __bf16 f2bf(float f) {
  unsigned u = __builtin_bit_cast(unsigned, f);
  u += 0x7FFFu + ((u >> 16) & 1u);               // round-to-nearest-even
  unsigned short h = (unsigned short)(u >> 16);
  return __builtin_bit_cast(__bf16, h);
}

// A-fragment (16x32 bf16), contiguous k order: lane half h holds k = 16h..16h+15
__device__ __forceinline__ v16bf load_a_frag(const __bf16* __restrict__ src, int ld) {
  const int lane = threadIdx.x & 31;
  const int m  = lane & 15;
  const int kb = (lane >> 4) << 4;   // 0 or 16
  v16bf a;
#pragma unroll
  for (int i = 0; i < 16; ++i) {
    a[i] = src[m * ld + kb + i];
  }
  return a;
}

// B-fragment (32x16 bf16) from an N-major (transposed) staging buffer
__device__ __forceinline__ v16bf load_bt_frag(const __bf16* __restrict__ srcT, int ld) {
  const int lane = threadIdx.x & 31;
  const int n  = lane & 15;
  const int kb = (lane >> 4) << 4;   // 0 or 16
  v16bf b;
#pragma unroll
  for (int i = 0; i < 16; ++i) b[i] = srcT[n * ld + kb + i];
  return b;
}

__device__ __forceinline__ v8f wmma_bf16(v16bf a, v16bf b, v8f c) {
  return __builtin_amdgcn_wmma_f32_16x16x32_bf16(false, a, false, b, (short)0, c, false, false);
}

// ---------------------------------------------------------------------------
// fp32 -> bf16 conversion (straight)
// ---------------------------------------------------------------------------
__global__ __launch_bounds__(256) void cvt_f32_bf16(const float* __restrict__ in,
                                                    __bf16* __restrict__ out, int n) {
  for (int i = blockIdx.x * blockDim.x + threadIdx.x; i < n; i += gridDim.x * blockDim.x)
    out[i] = f2bf(in[i]);
}

// ---------------------------------------------------------------------------
// fp32 (K,N) -> bf16 transposed (N,K), LDS-tiled so both sides coalesce.
// Pre-transposing the weights makes ALL GEMM LDS staging contiguous.
// ---------------------------------------------------------------------------
__global__ __launch_bounds__(256) void cvt_transpose_bf16(const float* __restrict__ in,
                                                          __bf16* __restrict__ out,
                                                          int K, int N) {
  __shared__ float tile[32][33];
  const int kt = blockIdx.y * 32, nt = blockIdx.x * 32;
  const int tx = threadIdx.x & 31, ty = threadIdx.x >> 5;   // 32 x 8
#pragma unroll
  for (int i = 0; i < 32; i += 8)
    tile[ty + i][tx] = in[(size_t)(kt + ty + i) * N + nt + tx];
  __syncthreads();
#pragma unroll
  for (int i = 0; i < 32; i += 8)
    out[(size_t)(nt + ty + i) * K + kt + tx] = f2bf(tile[tx][ty + i]);
}

// ---------------------------------------------------------------------------
// Tiled bf16 GEMM: C(M,N) = A(M,K) @ Bw(N,K)^T + bias, f32 accumulate.
// Block tile 128x128, 8 waves x (32x64). Double-buffered async LDS staging.
// mode 0: QKV epilogue (Q scaled 1/sqrt(dh) -> (B,nh,T,dh); K -> (B,nh,T,dh);
//         V -> transposed (B,nh,dh,T) so attention staging is contiguous)
// mode 1: f32 output row-major
// ---------------------------------------------------------------------------
__global__ __launch_bounds__(256) void gemm_bf16(
    const __bf16* __restrict__ A, const __bf16* __restrict__ Bw,
    const float* __restrict__ bias, int M, int N, int K, int mode,
    __bf16* __restrict__ qd, __bf16* __restrict__ kd, __bf16* __restrict__ vd,
    float* __restrict__ outf) {
  __shared__ __align__(16) __bf16 As[2][128 * 32];
  __shared__ __align__(16) __bf16 Bs[2][128 * 32];

  const int tid  = threadIdx.x;
  const int lane = tid & 31;
  const int w    = tid >> 5;
  const int wr   = w & 3;               // wave row: 4 x 32 rows
  const int wc   = w >> 2;              // wave col: 2 x 64 cols
  const int row0 = blockIdx.y * 128;
  const int col0 = blockIdx.x * 128;

  // staging coords: sr = output row/col index 0..127, 32B chunk per thread
  const int sr = tid >> 1;
  const int sh = (tid & 1) * 16;

  v8f acc[2][4];
#pragma unroll
  for (int i = 0; i < 2; ++i)
#pragma unroll
    for (int j = 0; j < 4; ++j)
#pragma unroll
      for (int r = 0; r < 8; ++r) acc[i][j][r] = 0.0f;

  const int nIter = K / 32;
  // prologue: stage tile 0 into buffer 0
  cp32B(A + (size_t)(row0 + sr) * K + 0 + sh, &As[0][sr * 32 + sh]);
  cp32B(Bw + (size_t)(col0 + sr) * K + 0 + sh, &Bs[0][sr * 32 + sh]);

  for (int it = 0; it < nIter; ++it) {
    const int cur = it & 1;
    if (it + 1 < nIter) {  // prefetch next tile into other buffer
      const int k0 = (it + 1) * 32;
      cp32B(A + (size_t)(row0 + sr) * K + k0 + sh, &As[cur ^ 1][sr * 32 + sh]);
      cp32B(Bw + (size_t)(col0 + sr) * K + k0 + sh, &Bs[cur ^ 1][sr * 32 + sh]);
      wait_async<4>();   // wait for current tile; next tile (4 ops) may stay in flight
    } else {
      wait_async<0>();
    }
    __syncthreads();

    v16bf af[2], bf[4];
#pragma unroll
    for (int rt = 0; rt < 2; ++rt) af[rt] = load_a_frag(&As[cur][(wr * 32 + rt * 16) * 32], 32);
#pragma unroll
    for (int ct = 0; ct < 4; ++ct) bf[ct] = load_bt_frag(&Bs[cur][(wc * 64 + ct * 16) * 32], 32);
#pragma unroll
    for (int rt = 0; rt < 2; ++rt)
#pragma unroll
      for (int ct = 0; ct < 4; ++ct) acc[rt][ct] = wmma_bf16(af[rt], bf[ct], acc[rt][ct]);
    __syncthreads();
  }

  // epilogue — C layout: VGPR r -> row r + 8*(lane>>4), col = lane&15
  const int half = lane >> 4;
  const int nloc = lane & 15;
#pragma unroll
  for (int rt = 0; rt < 2; ++rt)
#pragma unroll
    for (int ct = 0; ct < 4; ++ct)
#pragma unroll
      for (int r = 0; r < 8; ++r) {
        const int m = row0 + wr * 32 + rt * 16 + r + 8 * half;
        const int n = col0 + wc * 64 + ct * 16 + nloc;
        float v = acc[rt][ct][r] + bias[n];
        if (mode == 0) {
          const int b = m >> 11, t = m & (T_DIM - 1);
          const int which = n >> 10, c = n & (E_DIM - 1);
          const int h = c >> 6, d = c & (DH - 1);
          if (which == 0)
            qd[(((size_t)b * NH + h) * T_DIM + t) * DH + d] = f2bf(v * 0.125f);
          else if (which == 1)
            kd[(((size_t)b * NH + h) * T_DIM + t) * DH + d] = f2bf(v);
          else  // V stored transposed: (B, nh, dh, T)
            vd[(((size_t)b * NH + h) * DH + d) * T_DIM + t] = f2bf(v);
        } else {
          outf[(size_t)m * N + n] = v;
        }
      }
}

// ---------------------------------------------------------------------------
// Flash attention: one block per (128 q-rows, head, batch). 8 waves x 16 q-rows.
// Q,K in (B,nh,T,dh); V in (B,nh,dh,T) (pre-transposed). Q pre-scaled.
// Double-buffered async staging of K and V^T tiles. Output -> (B,T,E) bf16.
// ---------------------------------------------------------------------------
__global__ __launch_bounds__(256) void attn_flash(
    const __bf16* __restrict__ Q, const __bf16* __restrict__ Kb,
    const __bf16* __restrict__ Vt, __bf16* __restrict__ O) {
  __shared__ __align__(16) __bf16 Ks[2][128 * 64];    // [s][d]
  __shared__ __align__(16) __bf16 VTs[2][64 * 128];   // [d][s]
  __shared__ __align__(16) __bf16 Ps[8][16 * 128];    // per-wave P tile [m][s]

  const int tid  = threadIdx.x;
  const int lane = tid & 31;
  const int w    = tid >> 5;
  const int qblk = blockIdx.x, h = blockIdx.y, b = blockIdx.z;
  const size_t headoff = ((size_t)b * NH + h) * T_DIM * DH;   // same size both layouts
  const int q0 = qblk * 128 + w * 16;
  const int half = lane >> 4;
  const int nloc = lane & 15;

  // staging coords
  const int ks_s = tid >> 1, ks_d = (tid & 1) * 32;           // K tile: 64B per thread
  const int vt_d = tid >> 2, vt_s = (tid & 3) * 32;           // VT tile: 64B per thread

  // Q A-fragments straight from global (rows contiguous in d)
  const __bf16* qptr = Q + headoff + (size_t)q0 * DH;
  v16bf qf[2];
  qf[0] = load_a_frag(qptr, DH);
  qf[1] = load_a_frag(qptr + 32, DH);

  v8f o_acc[4];
#pragma unroll
  for (int ot = 0; ot < 4; ++ot)
#pragma unroll
    for (int r = 0; r < 8; ++r) o_acc[ot][r] = 0.0f;
  float m_i[8], l_i[8];
#pragma unroll
  for (int r = 0; r < 8; ++r) { m_i[r] = -3.0e38f; l_i[r] = 0.0f; }

  // prologue: stage kb=0 into buffer 0
  cp64B(Kb + headoff + (size_t)ks_s * DH + ks_d, &Ks[0][ks_s * DH + ks_d]);
  cp64B(Vt + headoff + (size_t)vt_d * T_DIM + vt_s, &VTs[0][vt_d * 128 + vt_s]);

  for (int kb = 0; kb <= qblk; ++kb) {
    const int cur = kb & 1;
    if (kb < qblk) {  // prefetch next K/V tile
      const int s0 = (kb + 1) * 128;
      cp64B(Kb + headoff + (size_t)(s0 + ks_s) * DH + ks_d, &Ks[cur ^ 1][ks_s * DH + ks_d]);
      cp64B(Vt + headoff + (size_t)vt_d * T_DIM + s0 + vt_s, &VTs[cur ^ 1][vt_d * 128 + vt_s]);
      wait_async<8>();
    } else {
      wait_async<0>();
    }
    __syncthreads();

    const bool diag = (kb == qblk);
    v8f s_acc[8];
#pragma unroll
    for (int ct = 0; ct < 8; ++ct) {
      if (diag && ct * 16 > w * 16 + 15) {   // fully above diagonal: skip WMMA
#pragma unroll
        for (int r = 0; r < 8; ++r) s_acc[ct][r] = MASKB;
        continue;
      }
#pragma unroll
      for (int r = 0; r < 8; ++r) s_acc[ct][r] = 0.0f;
#pragma unroll
      for (int ks = 0; ks < 2; ++ks) {
        v16bf kfrag = load_bt_frag(&Ks[cur][ct * 16 * DH + ks * 32], DH);
        s_acc[ct] = wmma_bf16(qf[ks], kfrag, s_acc[ct]);
      }
      if (diag) {  // partial causal mask (replace, matching reference)
        const int ng = kb * 128 + ct * 16 + nloc;
#pragma unroll
        for (int r = 0; r < 8; ++r) {
          const int mg = q0 + r + 8 * half;
          if (ng > mg) s_acc[ct][r] = MASKB;
        }
      }
    }

    // online softmax: row stats redundant across the 16 lanes of each half
    float mnew[8];
#pragma unroll
    for (int r = 0; r < 8; ++r) {
      float mx = s_acc[0][r];
#pragma unroll
      for (int ct = 1; ct < 8; ++ct) mx = fmaxf(mx, s_acc[ct][r]);
#pragma unroll
      for (int off = 1; off < 16; off <<= 1) mx = fmaxf(mx, __shfl_xor(mx, off, 32));
      mnew[r] = fmaxf(m_i[r], mx);
    }
    float rs[8];
#pragma unroll
    for (int r = 0; r < 8; ++r) {
      const float sc = __expf(m_i[r] - mnew[r]);
#pragma unroll
      for (int ot = 0; ot < 4; ++ot) o_acc[ot][r] *= sc;
      l_i[r] *= sc;
      rs[r] = 0.0f;
    }
#pragma unroll
    for (int ct = 0; ct < 8; ++ct)
#pragma unroll
      for (int r = 0; r < 8; ++r) {
        const float p = __expf(s_acc[ct][r] - mnew[r]);
        rs[r] += p;
        Ps[w][(r + 8 * half) * 128 + ct * 16 + nloc] = f2bf(p);
      }
#pragma unroll
    for (int r = 0; r < 8; ++r) {
      float t = rs[r];
#pragma unroll
      for (int off = 1; off < 16; off <<= 1) t += __shfl_xor(t, off, 32);
      l_i[r] += t;
      m_i[r] = mnew[r];
    }

    // O += P @ V   (same-wave LDS ops are in-order: no barrier needed for Ps)
#pragma unroll
    for (int ks2 = 0; ks2 < 4; ++ks2) {
      v16bf pa = load_a_frag(&Ps[w][ks2 * 32], 128);
#pragma unroll
      for (int ot = 0; ot < 4; ++ot) {
        v16bf vfrag = load_bt_frag(&VTs[cur][ot * 16 * 128 + ks2 * 32], 128);
        o_acc[ot] = wmma_bf16(pa, vfrag, o_acc[ot]);
      }
    }
    __syncthreads();
  }

  // normalize + write to (B, T, E) bf16 (E index = h*64 + d)
#pragma unroll
  for (int r = 0; r < 8; ++r) {
    const float rinv = 1.0f / l_i[r];
    const int t = q0 + r + 8 * half;
#pragma unroll
    for (int ot = 0; ot < 4; ++ot) {
      const int d = ot * 16 + nloc;
      O[((size_t)b * T_DIM + t) * E_DIM + h * DH + d] = f2bf(o_acc[ot][r] * rinv);
    }
  }
}

// ---------------------------------------------------------------------------
// Launcher — ws layout (bytes):
//   x_bf 0..8M | wAT_bf 8M..14M | wPT_bf 14M..16M | q 16M..24M | k 24M..32M
//   vT 32M..40M | attn 40M..48M
// ---------------------------------------------------------------------------
extern "C" void kernel_launch(void* const* d_in, const int* in_sizes, int n_in,
                              void* d_out, int out_size, void* d_ws, size_t ws_size,
                              hipStream_t stream) {
  const float* stacked = (const float*)d_in[0];
  const float* w_attn  = (const float*)d_in[1];
  const float* b_attn  = (const float*)d_in[2];
  const float* w_proj  = (const float*)d_in[3];
  const float* b_proj  = (const float*)d_in[4];
  float* out = (float*)d_out;
  char* ws = (char*)d_ws;

  __bf16* x_bf   = (__bf16*)(ws);
  __bf16* wAT_bf = (__bf16*)(ws + 8388608);    // (3072, 1024) = w_attn^T
  __bf16* wPT_bf = (__bf16*)(ws + 14680064);   // (1024, 1024) = w_proj^T
  __bf16* q_bf   = (__bf16*)(ws + 16777216);   // (B, nh, T, dh)
  __bf16* k_bf   = (__bf16*)(ws + 25165824);   // (B, nh, T, dh)
  __bf16* vT_bf  = (__bf16*)(ws + 33554432);   // (B, nh, dh, T)
  __bf16* a_bf   = (__bf16*)(ws + 41943040);   // (B, T, E)

  cvt_f32_bf16<<<1024, 256, 0, stream>>>(stacked, x_bf, M_ROWS * E_DIM);
  {
    dim3 g(E3 / 32, E_DIM / 32);
    cvt_transpose_bf16<<<g, 256, 0, stream>>>(w_attn, wAT_bf, E_DIM, E3);
  }
  {
    dim3 g(E_DIM / 32, E_DIM / 32);
    cvt_transpose_bf16<<<g, 256, 0, stream>>>(w_proj, wPT_bf, E_DIM, E_DIM);
  }

  dim3 g1(E3 / 128, M_ROWS / 128);
  gemm_bf16<<<g1, 256, 0, stream>>>(x_bf, wAT_bf, b_attn, M_ROWS, E3, E_DIM, 0,
                                    q_bf, k_bf, vT_bf, nullptr);

  dim3 g2(T_DIM / 128, NH, B_DIM);
  attn_flash<<<g2, 256, 0, stream>>>(q_bf, k_bf, vT_bf, a_bf);

  dim3 g3(E_DIM / 128, M_ROWS / 128);
  gemm_bf16<<<g3, 256, 0, stream>>>(a_bf, wPT_bf, b_proj, M_ROWS, E_DIM, E_DIM, 1,
                                    nullptr, nullptr, nullptr, out);
}
